// FragmentDecoder_83786222011215
// MI455X (gfx1250) — matrix operand order repaired and checked by reading, not compiled, not run on hardware
//
#include <hip/hip_runtime.h>

// FragmentDecoder for MI455X (gfx1250, wave32). All dense linear layers run on
// v_wmma_f32_16x16x32_f16 (f16 in, f32 accumulate). Edge-vector branch uses the
// rank-1 structure of edge_vec = w (outer) unit to avoid [E,64,3] GEMMs.

typedef __attribute__((ext_vector_type(16))) _Float16 v16h;
typedef __attribute__((ext_vector_type(8)))  float    v8f;
typedef __attribute__((ext_vector_type(2)))  _Float16 h2;

namespace {

constexpr int kNQ = 8192;
constexpr int kNC = 16384;
constexpr int kE  = 262144;
constexpr float kPI  = 3.14159265358979323846f;
constexpr float kCUT = 10.0f;
constexpr float kEPS = 1e-6f;

union Frag { v16h v; h2 p[8]; };

// ---- WMMA fragment loaders (per CDNA5 ISA layouts) -------------

// A: 16x32 f16 tile from f32 LDS [16 rows][lda cols]; lane(m=lane&15,g=lane>>4)
// VGPR j<4: K = kb + 8g + 2j ; j>=4: K = kb + 16 + 8g + 2(j-4)
__device__ __forceinline__ v16h load_a_frag(const float* As, int lda, int kb, int K) {
  const int lane = threadIdx.x & 31;
  const int m = lane & 15, g = lane >> 4;
  Frag a;
#pragma unroll
  for (int j = 0; j < 8; ++j) {
    int k0 = kb + ((j < 4) ? (8 * g + 2 * j) : (16 + 8 * g + 2 * (j - 4)));
    float f0 = (k0     < K) ? As[m * lda + k0    ] : 0.0f;
    float f1 = (k0 + 1 < K) ? As[m * lda + k0 + 1] : 0.0f;
    h2 t;
    t.x = (_Float16)f0;
    t.y = (_Float16)f1;
    a.p[j] = t;
  }
  return a.v;
}

// B: 32x16 tile; computes x @ W^T so B[k][n] = W[(ob+n)][koff + kb + k].
// lane(n=lane&15,g=lane>>4); VGPR j<4: K = kb + 8g + 2j ; j>=4: K = kb + 16 + 8g + 2(j-4)
__device__ __forceinline__ v16h load_b_frag(const float* W, int ldw, int koff,
                                            int kb, int K, int ob, int O) {
  const int lane = threadIdx.x & 31;
  const int n = lane & 15, g = lane >> 4;
  const bool ok = (ob + n) < O;
  const float* wr = W + (size_t)(ok ? (ob + n) : 0) * ldw + koff;
  Frag b;
#pragma unroll
  for (int j = 0; j < 8; ++j) {
    int k0 = kb + ((j < 4) ? (8 * g + 2 * j) : (16 + 8 * g + 2 * (j - 4)));
    h2 t;
    t.x = (_Float16)((ok && k0     < K) ? wr[k0]     : 0.0f);
    t.y = (_Float16)((ok && k0 + 1 < K) ? wr[k0 + 1] : 0.0f);
    b.p[j] = t;
  }
  return b.v;
}

__device__ __forceinline__ v8f wmma_kloop(const float* As, int lda, const float* W,
                                          int ldw, int koff, int K, int ob, int O, v8f acc) {
  for (int kb = 0; kb < K; kb += 32) {
    v16h a = load_a_frag(As, lda, kb, K);
    v16h b = load_b_frag(W, ldw, koff, kb, K, ob, O);
    acc = __builtin_amdgcn_wmma_f32_16x16x32_f16(false, a, false, b, (short)0, acc,
                                                 false, false);
  }
  return acc;
}

__device__ __forceinline__ void gemm_epilogue(float* Out, int ldo, int o, int O,
                                              const float* bias, int act, int g, v8f c) {
  if (o < O) {
    float bv = bias ? bias[o] : 0.0f;
#pragma unroll
    for (int r = 0; r < 8; ++r) {
      float v = c[r] + bv;
      if (act == 1) v = 1.0f / (1.0f + expf(-v));
      else if (act == 2) v = (v > 0.0f) ? v : 0.01f * v;
      Out[(r + 8 * g) * ldo + o] = v;
    }
  }
}

// Out[16][O] = A[16][K] @ W[:,koff:koff+K]^T (+bias, +activation).
// 4 waves; O in {64,128}. For O=128 each wave owns two column tiles and each A
// fragment is loaded once per kb. ACT: 0 none, 1 sigmoid, 2 leaky(0.01)
__device__ __forceinline__ void block_gemm(const float* As, int lda, const float* W,
                                           int ldw, int koff, int K, int O,
                                           float* Out, int ldo, const float* bias, int act) {
  const int ob0 = (threadIdx.x >> 5) * 16;
  const bool two = (O > 64);
  v8f c0 = {}, c1 = {};
  for (int kb = 0; kb < K; kb += 32) {
    v16h a = load_a_frag(As, lda, kb, K);
    v16h b0 = load_b_frag(W, ldw, koff, kb, K, ob0, O);
    c0 = __builtin_amdgcn_wmma_f32_16x16x32_f16(false, a, false, b0, (short)0, c0,
                                                false, false);
    if (two) {
      v16h b1 = load_b_frag(W, ldw, koff, kb, K, ob0 + 64, O);
      c1 = __builtin_amdgcn_wmma_f32_16x16x32_f16(false, a, false, b1, (short)0, c1,
                                                  false, false);
    }
  }
  const int lane = threadIdx.x & 31;
  const int n = lane & 15, g = lane >> 4;
  gemm_epilogue(Out, ldo, ob0 + n, O, bias, act, g, c0);
  if (two) gemm_epilogue(Out, ldo, ob0 + 64 + n, O, bias, act, g, c1);
}

// Three GEMMs sharing one weight matrix (K=64, O=64): Out[p] = A[p] @ W^T,
// planes at +p*1024. B fragments are loaded once per wave.
__device__ __forceinline__ void block_gemm3(const float* As3, const float* W, float* Out3) {
  const int ob = (threadIdx.x >> 5) * 16;
  const v16h b0 = load_b_frag(W, 64, 0, 0, 64, ob, 64);
  const v16h b1 = load_b_frag(W, 64, 0, 32, 64, ob, 64);
  const int lane = threadIdx.x & 31;
  const int n = lane & 15, g = lane >> 4;
#pragma unroll
  for (int p = 0; p < 3; ++p) {
    v16h a0 = load_a_frag(As3 + p * 1024, 64, 0, 64);
    v16h a1 = load_a_frag(As3 + p * 1024, 64, 32, 64);
    v8f c = {};
    c = __builtin_amdgcn_wmma_f32_16x16x32_f16(false, a0, false, b0, (short)0, c, false, false);
    c = __builtin_amdgcn_wmma_f32_16x16x32_f16(false, a1, false, b1, (short)0, c, false, false);
#pragma unroll
    for (int r = 0; r < 8; ++r) Out3[p * 1024 + (r + 8 * g) * 64 + ob + n] = c[r];
  }
}

// ---- gvlinear weights -----------------------------------------------------
struct GVW {
  const float* W1;    // lin_vec1 [64,64]
  const float* W2;    // lin_vec2 [64,64]
  const float* Lsca;  // lin_sca  [os, Ksca]
  const float* Gw;    // gate_w   [64, os]
  const float* Gb;    // gate_b   [64]
};

// gvlinear(128,64 -> 128,64). S* are [16][128] LDS, V*/ITMP are 3 planes of
// [16][64] LDS (k*1024+m*64+c), CAT is [16][192], GT is [16][64]. 128 threads.
__device__ void gvlinear128(const GVW w, const float* Sin, const float* Vin,
                            float* Sout, float* Vout, float* ITMP, float* CAT, float* GT) {
  const int t = threadIdx.x;
  block_gemm3(Vin, w.W1, ITMP);
  __syncthreads();
  for (int idx = t; idx < 1024; idx += 128) {
    int m = idx >> 6;
    float x = ITMP[idx], y = ITMP[1024 + idx], z = ITMP[2048 + idx];
    CAT[m * 192 + (idx & 63)] = sqrtf(x * x + y * y + z * z);
  }
  for (int idx = t; idx < 2048; idx += 128) {
    int m = idx >> 7, j = idx & 127;
    CAT[m * 192 + 64 + j] = Sin[m * 128 + j];
  }
  __syncthreads();
  block_gemm(CAT, 192, w.Lsca, 192, 0, 192, 128, Sout, 128, nullptr, 0);
  __syncthreads();
  block_gemm(Sout, 128, w.Gw, 128, 0, 128, 64, GT, 64, w.Gb, 1);
  __syncthreads();
  block_gemm3(ITMP, w.W2, Vout);
  __syncthreads();
  for (int idx = t; idx < 3072; idx += 128) {
    int r = idx & 1023, m = r >> 6, o = r & 63;
    Vout[idx] *= GT[m * 64 + o];
  }
  __syncthreads();
}

// ---- Kernel: per-launch precompute ---------------------------------------
// a = W1_e @ vec_exp_w ; b1 = Lsca_e[:, :64] @ |a| ; c2 = W2_e @ a ;
// num_gen = min(max(num_gen_atoms), 12)
__global__ void k_prep(const float* W1e, const float* W2e, const float* LscaE,
                       const float* wexp, const int* ngen_in,
                       float* b1, float* c2, int* ngen_out) {
  __shared__ float a_sh[64];
  const int t = threadIdx.x;
  if (t < 64) {
    float s = 0.0f;
    for (int c = 0; c < 64; ++c) s += W1e[t * 64 + c] * wexp[c];
    a_sh[t] = s;
  }
  __syncthreads();
  if (t < 64) {
    float s2 = 0.0f, s1 = 0.0f;
    for (int h = 0; h < 64; ++h) {
      s2 += W2e[t * 64 + h] * a_sh[h];
      s1 += LscaE[t * 128 + h] * fabsf(a_sh[h]);
    }
    c2[t] = s2;
    b1[t] = s1;
  }
  if (t == 0) {
    int mx = 0;
    for (int i = 0; i < 16; ++i) { int v = ngen_in[i]; mx = v > mx ? v : mx; }
    ngen_out[0] = mx < 12 ? mx : 12;
  }
}

// ---- Kernel: node_gv over compose nodes ----------------------------------
__global__ void __launch_bounds__(128) k_node(const float* sca_g, const float* vec_g,
                                              GVW w, float* ns_g, float* nv_g) {
  __shared__ float V[3072];    // 3 x [16][64]
  __shared__ float IT[3072];
  __shared__ float CAT[16 * 320];
  __shared__ float OS[2048];
  __shared__ float GT[1024];
  const int t = threadIdx.x;
  const int n0 = blockIdx.x * 16;
  for (int idx = t; idx < 3072; idx += 128) {
    int k = idx >> 10, r = idx & 1023, m = r >> 6, c = r & 63;
    V[idx] = vec_g[((size_t)(n0 + m) * 64 + c) * 3 + k];
  }
  for (int idx = t; idx < 16 * 256; idx += 128) {
    int m = idx >> 8, c = idx & 255;
    CAT[m * 320 + 64 + c] = sca_g[(size_t)(n0 + m) * 256 + c];
  }
  __syncthreads();
  block_gemm3(V, w.W1, IT);
  __syncthreads();
  for (int idx = t; idx < 1024; idx += 128) {
    int m = idx >> 6, h = idx & 63;
    float x = IT[idx], y = IT[1024 + idx], z = IT[2048 + idx];
    CAT[m * 320 + h] = sqrtf(x * x + y * y + z * z);
  }
  __syncthreads();
  block_gemm(CAT, 320, w.Lsca, 320, 0, 320, 128, OS, 128, nullptr, 0);
  __syncthreads();
  block_gemm(OS, 128, w.Gw, 128, 0, 128, 64, GT, 64, w.Gb, 1);
  __syncthreads();
  for (int idx = t; idx < 2048; idx += 128) {
    int m = idx >> 7, o = idx & 127;
    ns_g[(size_t)(n0 + m) * 128 + o] = OS[idx];
  }
  // gated out_vec straight to global (one 16x16 tile per wave, O=64; B hoisted)
  {
    const int ob = (t >> 5) * 16;
    const v16h b0 = load_b_frag(w.W2, 64, 0, 0, 64, ob, 64);
    const v16h b1 = load_b_frag(w.W2, 64, 0, 32, 64, ob, 64);
    const int lane = t & 31, n = lane & 15, g = lane >> 4;
    const int o = ob + n;
#pragma unroll
    for (int k = 0; k < 3; ++k) {
      v16h a0 = load_a_frag(IT + k * 1024, 64, 0, 64);
      v16h a1 = load_a_frag(IT + k * 1024, 64, 32, 64);
      v8f c = {};
      c = __builtin_amdgcn_wmma_f32_16x16x32_f16(false, a0, false, b0, (short)0, c, false, false);
      c = __builtin_amdgcn_wmma_f32_16x16x32_f16(false, a1, false, b1, (short)0, c, false, false);
#pragma unroll
      for (int r = 0; r < 8; ++r) {
        int m = r + 8 * g;
        nv_g[((size_t)(n0 + m) * 64 + o) * 3 + k] = GT[m * 64 + o] * c[r];
      }
    }
  }
}

// ---- Kernel: per-edge message + atomic segment sum -----------------------
struct EdgeP {
  const float* b1; const float* c2;                        // prep vectors [64]
  const float* e_Lsca; const float* e_Gw; const float* e_Gb; const float* e_dir;
  const float* sca_w; const float* sca_b;                  // [128,64],[128]
  const float* e2n_w; const float* e2n_b;                  // [64,64],[64]
  const float* n2e_w; const float* n2e_b;                  // [64,128],[64]
  const float* edge_vn;                                    // [64,64]
  GVW out;                                                 // out_gv (Ksca=192)
};

__global__ void __launch_bounds__(128) k_edge(const int* eidx, const float* posq,
                                              const float* posc, const float* ns_g,
                                              const float* nv_g, EdgeP P,
                                              float* hidS, float* hidV) {
  __shared__ float NS[2048];   // gathered ns ; later out_sca
  __shared__ float NVC[3072];  // gathered nv (3 planes) ; later CAT[16][192]
  __shared__ float IT[3072];
  __shared__ float Y[2048];
  __shared__ float B1[1024], B2[1024], B3[1024], B4[1024], B5[1024];
  __shared__ float U[16][3], UN[16], U2[16], CC[16], DIST[16];
  __shared__ int R0[16], R1[16];
  const int t = threadIdx.x;
  const int e0 = blockIdx.x * 16;
  if (t < 16) {
    int e = e0 + t;
    int iq = eidx[e], ic = eidx[kE + e];
    R0[t] = iq; R1[t] = ic;
    float vx = posq[(size_t)iq * 3 + 0] - posc[(size_t)ic * 3 + 0];
    float vy = posq[(size_t)iq * 3 + 1] - posc[(size_t)ic * 3 + 1];
    float vz = posq[(size_t)iq * 3 + 2] - posc[(size_t)ic * 3 + 2];
    float d = sqrtf(vx * vx + vy * vy + vz * vz);
    float inv = 1.0f / (d + 1e-7f);
    U[t][0] = vx * inv; U[t][1] = vy * inv; U[t][2] = vz * inv;
    float un = d * inv;
    UN[t] = un; U2[t] = un * un; DIST[t] = d;
    CC[t] = (d <= kCUT) ? 0.5f * (cosf(d * kPI / kCUT) + 1.0f) : 0.0f;
  }
  __syncthreads();
  for (int idx = t; idx < 2048; idx += 128) {
    int m = idx >> 7, o = idx & 127;
    NS[idx] = ns_g[(size_t)R1[m] * 128 + o];
  }
  for (int idx = t; idx < 3072; idx += 128) {
    int k = idx >> 10, r = idx & 1023, m = r >> 6, c = r & 63;
    NVC[idx] = nv_g[((size_t)R1[m] * 64 + c) * 3 + k];
  }
  const float stepg = kCUT / 63.0f;
  const float coeff = -0.5f / (stepg * stepg);
  for (int idx = t; idx < 1024; idx += 128) {      // Gaussian smearing
    int m = idx >> 6, c = idx & 63;
    float dd = DIST[m] - stepg * (float)c;
    B1[idx] = expf(coeff * dd * dd);
  }
  __syncthreads();
  // edge gvlinear scalar path: os_e = un*b1 + edge_sca @ Lsca_e[:,64:128]^T
  block_gemm(B1, 64, P.e_Lsca, 128, 64, 64, 64, B2, 64, nullptr, 0);
  __syncthreads();
  for (int idx = t; idx < 1024; idx += 128) B2[idx] += UN[idx >> 6] * P.b1[idx & 63];
  __syncthreads();
  block_gemm(B2, 64, P.e_Gw, 64, 0, 64, 64, B3, 64, P.e_Gb, 1);  // gate_e
  __syncthreads();
  for (int idx = t; idx < 1024; idx += 128) {      // es = leaky(os); g2 = gate*c2
    float v = B2[idx];
    B2[idx] = (v > 0.0f) ? v : 0.01f * v;
    B3[idx] *= P.c2[idx & 63];
  }
  __syncthreads();
  block_gemm(B3, 64, P.e_dir, 64, 0, 64, 64, B1, 64, nullptr, 0); // dvec
  __syncthreads();
  for (int idx = t; idx < 1024; idx += 128) {      // vn_leaky coefficient h2
    float g2 = B3[idx], dv = B1[idx], u2 = U2[idx >> 6];
    float dot = g2 * dv * u2, dsq = dv * dv * u2;
    float pv = g2 - (dot / (dsq + kEPS)) * dv;
    B3[idx] = 0.2f * g2 + 0.8f * ((dot >= 0.0f) ? g2 : pv);
  }
  __syncthreads();
  block_gemm(B3, 64, P.edge_vn, 64, 0, 64, 64, B1, 64, nullptr, 0);        // q
  block_gemm(B2, 64, P.e2n_w, 64, 0, 64, 64, B4, 64, P.e2n_b, 0);          // alpha
  block_gemm(B2, 64, P.sca_w, 64, 0, 64, 128, Y, 128, P.sca_b, 0);         // sca path
  block_gemm(NS, 128, P.n2e_w, 128, 0, 128, 64, B5, 64, P.n2e_b, 0);       // beta
  __syncthreads();
  for (int idx = t; idx < 2048; idx += 128) Y[idx] *= NS[idx];    // y_sca
  for (int idx = t; idx < 1024; idx += 128) B5[idx] *= B1[idx];   // beta*q
  __syncthreads();
  block_gemm(B5, 64, P.out.W1, 64, 0, 64, 64, B1, 64, nullptr, 0); // s1 (rank-1)
  __syncthreads();
  // inter[k] = (alpha*nv) @ W1o^T : scale nv planes in place, shared-B triple
  for (int idx = t; idx < 3072; idx += 128) NVC[idx] *= B4[idx & 1023];
  __syncthreads();
  block_gemm3(NVC, P.out.W1, IT);
  __syncthreads();
  for (int idx = t; idx < 3072; idx += 128) {      // + s1 * unit[k]
    int k = idx >> 10, r = idx & 1023, m = r >> 6;
    IT[idx] += B1[r] * U[m][k];
  }
  __syncthreads();
  for (int idx = t; idx < 1024; idx += 128) {      // CAT = [norm | y_sca]
    int m = idx >> 6, h = idx & 63;
    float x = IT[idx], y = IT[1024 + idx], z = IT[2048 + idx];
    NVC[m * 192 + h] = sqrtf(x * x + y * y + z * z);
  }
  for (int idx = t; idx < 2048; idx += 128) {
    int m = idx >> 7, j = idx & 127;
    NVC[m * 192 + 64 + j] = Y[m * 128 + j];
  }
  __syncthreads();
  block_gemm(NVC, 192, P.out.Lsca, 192, 0, 192, 128, NS, 128, nullptr, 0); // out_sca
  __syncthreads();
  block_gemm(NS, 128, P.out.Gw, 128, 0, 128, 64, B2, 64, P.out.Gb, 1);     // gate_o
  __syncthreads();
  for (int idx = t; idx < 2048; idx += 128) {      // m_sca segment-sum
    int m = idx >> 7, o = idx & 127;
    atomicAdd(&hidS[(size_t)R0[m] * 128 + o], NS[idx] * CC[m]);
  }
  {                                                // m_vec = gate*outvec*C
    const int ob = (t >> 5) * 16;
    const v16h b0 = load_b_frag(P.out.W2, 64, 0, 0, 64, ob, 64);
    const v16h b1 = load_b_frag(P.out.W2, 64, 0, 32, 64, ob, 64);
    const int lane = t & 31, n = lane & 15, g = lane >> 4;
    const int o = ob + n;
#pragma unroll
    for (int k = 0; k < 3; ++k) {
      v16h a0 = load_a_frag(IT + k * 1024, 64, 0, 64);
      v16h a1 = load_a_frag(IT + k * 1024, 64, 32, 64);
      v8f c = {};
      c = __builtin_amdgcn_wmma_f32_16x16x32_f16(false, a0, false, b0, (short)0, c, false, false);
      c = __builtin_amdgcn_wmma_f32_16x16x32_f16(false, a1, false, b1, (short)0, c, false, false);
#pragma unroll
      for (int r = 0; r < 8; ++r) {
        int m = r + 8 * g;
        atomicAdd(&hidV[((size_t)R0[m] * 64 + o) * 3 + k], B2[m * 64 + o] * CC[m] * c[r]);
      }
    }
  }
}

// ---- Kernel: one scan step -----------------------------------------------
struct StepP {
  GVW gvp; const float* gvp_dir;
  GVW gv;
  const float* on_W1; const float* on_Lsca; const float* on_W2;
  const float* on_gw; const float* on_gb;
};

__global__ void __launch_bounds__(128) k_step(int step, const float* Sin_g,
                                              const float* Vin_g, float* Sout_g,
                                              float* Vout_g, StepP P, const float* posq,
                                              const int* ngen_p, float* outEl,
                                              float* outPos) {
  __shared__ float S[2048], S2[2048];
  __shared__ float V[3072], V2[3072], IT[3072], CAT[3072];
  __shared__ float GT[1024];
  __shared__ float ES9[144];
  __shared__ float G1[16];
  const int t = threadIdx.x;
  const int q0 = blockIdx.x * 16;
  for (int idx = t; idx < 2048; idx += 128) {
    int m = idx >> 7, j = idx & 127;
    S[idx] = Sin_g[(size_t)(q0 + m) * 128 + j];
  }
  for (int idx = t; idx < 3072; idx += 128) {
    int k = idx >> 10, r = idx & 1023, m = r >> 6, c = r & 63;
    V[idx] = Vin_g[((size_t)(q0 + m) * 64 + c) * 3 + k];
  }
  __syncthreads();
  gvlinear128(P.gvp, S, V, S2, V2, IT, CAT, GT);
  for (int idx = t; idx < 2048; idx += 128) {      // leaky 0.01 on sca
    float v = S2[idx];
    S2[idx] = (v > 0.0f) ? v : 0.01f * v;
  }
  __syncthreads();
  block_gemm3(V2, P.gvp_dir, IT);                  // d = v @ dir^T
  __syncthreads();
  for (int idx = t; idx < 1024; idx += 128) {      // vn_leaky_relu (slope 0.2)
    float x0 = V2[idx], x1 = V2[1024 + idx], x2 = V2[2048 + idx];
    float d0 = IT[idx], d1 = IT[1024 + idx], d2 = IT[2048 + idx];
    float dot = x0 * d0 + x1 * d1 + x2 * d2;
    float dsq = d0 * d0 + d1 * d1 + d2 * d2;
    float f = dot / (dsq + kEPS);
    bool pos = (dot >= 0.0f);
    V2[idx]        = 0.2f * x0 + 0.8f * (pos ? x0 : (x0 - f * d0));
    V2[1024 + idx] = 0.2f * x1 + 0.8f * (pos ? x1 : (x1 - f * d1));
    V2[2048 + idx] = 0.2f * x2 + 0.8f * (pos ? x2 : (x2 - f * d2));
  }
  __syncthreads();
  gvlinear128(P.gv, S2, V2, S, V, IT, CAT, GT);    // carry update
  for (int idx = t; idx < 2048; idx += 128) {
    int m = idx >> 7, j = idx & 127;
    Sout_g[(size_t)(q0 + m) * 128 + j] = S[idx];
  }
  for (int idx = t; idx < 3072; idx += 128) {
    int k = idx >> 10, r = idx & 1023, m = r >> 6, c = r & 63;
    Vout_g[((size_t)(q0 + m) * 64 + c) * 3 + k] = V[idx];
  }
  // out_net: gvlinear(128,64 -> 9,1)
  block_gemm3(V, P.on_W1, IT);
  __syncthreads();
  for (int idx = t; idx < 1024; idx += 128) {
    int m = idx >> 6, h = idx & 63;
    float x = IT[idx], y = IT[1024 + idx], z = IT[2048 + idx];
    CAT[m * 192 + h] = sqrtf(x * x + y * y + z * z);
  }
  for (int idx = t; idx < 2048; idx += 128) {
    int m = idx >> 7, j = idx & 127;
    CAT[m * 192 + 64 + j] = S[idx];
  }
  __syncthreads();
  if ((t >> 5) == 0) {                             // e_sca: one tile, O=9 guarded
    v8f c = {};
    c = wmma_kloop(CAT, 192, P.on_Lsca, 192, 0, 192, 0, 9, c);
    const int lane = t & 31, n = lane & 15, g = lane >> 4;
    if (n < 9) {
#pragma unroll
      for (int r = 0; r < 8; ++r) ES9[(r + 8 * g) * 9 + n] = c[r];
    }
  }
  __syncthreads();
  if (t < 16) {                                    // scalar gate (ov=1)
    float acc = P.on_gb[0];
    for (int o = 0; o < 9; ++o) acc += ES9[t * 9 + o] * P.on_gw[o];
    G1[t] = 1.0f / (1.0f + expf(-acc));
  }
  __syncthreads();
  const int ng = ngen_p[0];
  const bool valid = (step <= ng);
  for (int idx = t; idx < 144; idx += 128) {
    int m = idx / 9, o = idx % 9;
    outEl[((size_t)(q0 + m) * 13 + step) * 9 + o] = valid ? ES9[idx] : 0.0f;
  }
  if (step < 12 && t < 48) {                       // e_vec (ov=1) + pos
    int m = t / 3, k = t % 3;
    float acc = 0.0f;
    for (int h = 0; h < 64; ++h) acc += P.on_W2[h] * IT[k * 1024 + m * 64 + h];
    float val = valid ? (posq[(size_t)(q0 + m) * 3 + k] + G1[m] * acc) : 0.0f;
    outPos[((size_t)(q0 + m) * 12 + step) * 3 + k] = val;
  }
}

}  // namespace

extern "C" void kernel_launch(void* const* d_in, const int* in_sizes, int n_in,
                              void* d_out, int out_size, void* d_ws, size_t ws_size,
                              hipStream_t stream) {
  (void)in_sizes; (void)n_in; (void)out_size; (void)ws_size;
  // Top-level inputs in setup_inputs() insertion order; 'params' flattened as a
  // JAX pytree (dict keys sorted recursively).
  const float* h_sca = (const float*)d_in[0];
  const float* h_vec = (const float*)d_in[1];
  const float* posq  = (const float*)d_in[2];
  const float* posc  = (const float*)d_in[3];
  const int*   eidx  = (const int*)d_in[4];
  const int*   ngen_in = (const int*)d_in[5];
  int a = 6;
  const float* msg_e2n_b   = (const float*)d_in[a++];
  const float* msg_e2n_w   = (const float*)d_in[a++];
  const float* egvp_dir    = (const float*)d_in[a++];
  const float* egv_gb      = (const float*)d_in[a++];
  const float* egv_gw      = (const float*)d_in[a++];
  const float* egv_lsca    = (const float*)d_in[a++];
  const float* egv_w1      = (const float*)d_in[a++];
  const float* egv_w2      = (const float*)d_in[a++];
  const float* msg_edge_vn = (const float*)d_in[a++];
  const float* msg_n2e_b   = (const float*)d_in[a++];
  const float* msg_n2e_w   = (const float*)d_in[a++];
  const float* ngv_gb      = (const float*)d_in[a++];
  const float* ngv_gw      = (const float*)d_in[a++];
  const float* ngv_lsca    = (const float*)d_in[a++];
  const float* ngv_w1      = (const float*)d_in[a++];
  const float* ngv_w2      = (const float*)d_in[a++];
  const float* ogv_gb      = (const float*)d_in[a++];
  const float* ogv_gw      = (const float*)d_in[a++];
  const float* ogv_lsca    = (const float*)d_in[a++];
  const float* ogv_w1      = (const float*)d_in[a++];
  const float* ogv_w2      = (const float*)d_in[a++];
  const float* msg_sca_b   = (const float*)d_in[a++];
  const float* msg_sca_w   = (const float*)d_in[a++];
  const float* on_gb       = (const float*)d_in[a++];
  const float* on_gw       = (const float*)d_in[a++];
  const float* on_lsca     = (const float*)d_in[a++];
  const float* on_w1       = (const float*)d_in[a++];
  const float* on_w2       = (const float*)d_in[a++];
  const float* pgv_gb      = (const float*)d_in[a++];
  const float* pgv_gw      = (const float*)d_in[a++];
  const float* pgv_lsca    = (const float*)d_in[a++];
  const float* pgv_w1      = (const float*)d_in[a++];
  const float* pgv_w2      = (const float*)d_in[a++];
  const float* pgvp_dir    = (const float*)d_in[a++];
  const float* pgvp_gb     = (const float*)d_in[a++];
  const float* pgvp_gw     = (const float*)d_in[a++];
  const float* pgvp_lsca   = (const float*)d_in[a++];
  const float* pgvp_w1     = (const float*)d_in[a++];
  const float* pgvp_w2     = (const float*)d_in[a++];
  const float* vec_exp_w   = (const float*)d_in[a++];

  // Workspace layout (floats). ns/nv region is reused as the scan's ping
  // buffer after the edge phase completes (stream-ordered).
  float* ws   = (float*)d_ws;
  float* nsW  = ws;                                  // NC*128
  float* nvW  = nsW + (size_t)kNC * 128;             // NC*192
  float* hidS = nvW + (size_t)kNC * 192;             // NQ*128 (carry A sca)
  float* hidV = hidS + (size_t)kNQ * 128;            // NQ*192 (carry A vec)
  float* prep = hidV + (size_t)kNQ * 192;            // b1[64], c2[64]
  float* b1   = prep;
  float* c2   = prep + 64;
  int*   ngen_ws = (int*)(prep + 128);
  float* cbS = ws;                                   // carry B (reuses ns/nv)
  float* cbV = ws + (size_t)kNQ * 128;
  float* outEl  = (float*)d_out;                     // [NQ,13,9]
  float* outPos = outEl + (size_t)kNQ * 13 * 9;      // [NQ,12,3]

  hipMemsetAsync(hidS, 0, (size_t)(kNQ * 128 + kNQ * 192) * sizeof(float), stream);
  k_prep<<<1, 64, 0, stream>>>(egv_w1, egv_w2, egv_lsca, vec_exp_w, ngen_in,
                               b1, c2, ngen_ws);

  GVW nodeW{ngv_w1, ngv_w2, ngv_lsca, ngv_gw, ngv_gb};
  k_node<<<kNC / 16, 128, 0, stream>>>(h_sca, h_vec, nodeW, nsW, nvW);

  EdgeP ep{b1, c2,
           egv_lsca, egv_gw, egv_gb, egvp_dir,
           msg_sca_w, msg_sca_b,
           msg_e2n_w, msg_e2n_b,
           msg_n2e_w, msg_n2e_b,
           msg_edge_vn,
           GVW{ogv_w1, ogv_w2, ogv_lsca, ogv_gw, ogv_gb}};
  k_edge<<<kE / 16, 128, 0, stream>>>(eidx, posq, posc, nsW, nvW, ep, hidS, hidV);

  StepP sp{GVW{pgvp_w1, pgvp_w2, pgvp_lsca, pgvp_gw, pgvp_gb}, pgvp_dir,
           GVW{pgv_w1, pgv_w2, pgv_lsca, pgv_gw, pgv_gb},
           on_w1, on_lsca, on_w2, on_gw, on_gb};
  for (int s = 0; s <= 12; ++s) {
    const float *si, *vi;
    float *so, *vo;
    if ((s & 1) == 0) { si = hidS; vi = hidV; so = cbS; vo = cbV; }
    else              { si = cbS;  vi = cbV;  so = hidS; vo = hidV; }
    k_step<<<kNQ / 16, 128, 0, stream>>>(s, si, vi, so, vo, sp, posq, ngen_ws,
                                         outEl, outPos);
  }
}
